// EquAttention_11948599018113
// MI455X (gfx1250) — matrix operand order repaired and checked by reading, not compiled, not run on hardware
//
#include <hip/hip_runtime.h>

typedef __attribute__((ext_vector_type(16))) __bf16 v16bf;
typedef __attribute__((ext_vector_type(8)))  float  v8f;

#define N_ATOM 512
#define E_EDGE 8192
#define S_SPH  9
#define CIN    128
#define CH     256
#define NHEAD  8
#define DHEAD  32
#define FDIM   288   // S_SPH * DHEAD
#define NSEG   16
#define KSPLIT 8     // K-split factor for out_gemm (8192/8 = 1024 edges per job)

// A-matrix K index inside a 32-wide chunk (16-bit A 16x32 layout):
// lanes 0-15: VGPR0..3 -> K 0..7, VGPR4..7 -> K 16..23 ; lanes 16-31: +8
__device__ __forceinline__ int kA(int t, int laneHalf) {
    int v = t >> 1, j = t & 1;
    int base = (v < 4) ? (v * 2) : (16 + (v - 4) * 2);
    return base + j + laneHalf * 8;
}
// B-matrix K index: same order as A;
// lanes 0-15: K 0..7, 16..23 ; lanes 16-31: +8
__device__ __forceinline__ int kB(int t, int laneHalf) {
    return t + laneHalf * 8 + ((t >= 8) ? 8 : 0);
}

__device__ __forceinline__ unsigned fenc(float f) {
    unsigned u = __float_as_uint(f);
    return (u & 0x80000000u) ? ~u : (u | 0x80000000u);
}
__device__ __forceinline__ float fdec(unsigned u) {
    return (u & 0x80000000u) ? __uint_as_float(u & 0x7fffffffu) : __uint_as_float(~u);
}

// ---------------- so3_linear (input side): x (rows, S, CIN) -> bf16 (H, rows, 288)
__global__ void so3_in_linear(const float* __restrict__ x, const float* __restrict__ W,
                              const float* __restrict__ b, __bf16* __restrict__ outp,
                              float scale) {
    __shared__ float xr[CIN];
    int ns = blockIdx.x;               // n*S + s   (N_ATOM*S_SPH blocks)
    int o  = threadIdx.x;              // 0..255
    int n  = ns / S_SPH, s = ns % S_SPH;
    int l  = (s >= 4) ? 2 : ((s >= 1) ? 1 : 0);
    if (o < CIN) xr[o] = x[(size_t)ns * CIN + o];
    __syncthreads();
    const float* Wl = W + (size_t)l * CIN * CH;
    float acc = 0.0f;
    for (int i = 0; i < CIN; ++i) acc += xr[i] * Wl[i * CH + o];
    if (s == 0) acc += b[o];
    int h = o >> 5, dd = o & 31;
    outp[((size_t)(h * N_ATOM + n)) * FDIM + s * DHEAD + dd] = (__bf16)(acc * scale);
}

// ---------------- scores GEMM: per-wave 16(n) x 64(e) tile, K = 288
__global__ void scores_gemm(const __bf16* __restrict__ qh, const __bf16* __restrict__ kh,
                            const int* __restrict__ atom_index,
                            const int* __restrict__ edge_map_tab,
                            const float* __restrict__ attn_bias,
                            float* __restrict__ scores) {
    int w    = blockIdx.x * 8 + (threadIdx.x >> 5);   // 32768 wave jobs
    int h    = w >> 12;                               // / (32*128)
    int rem  = w & 4095;
    int mt   = rem >> 7;                              // n tile (0..31)
    int es   = rem & 127;                             // 64-wide e strip (0..127)
    int lane = threadIdx.x & 31;
    int lh   = lane >> 4;
    int l16  = lane & 15;

    const __bf16* Ap = qh + ((size_t)(h * N_ATOM + mt * 16 + l16)) * FDIM;
    int eb = es * 64;
    int atoms[4];
#pragma unroll
    for (int nt = 0; nt < 4; ++nt) atoms[nt] = atom_index[eb + nt * 16 + l16];

    v8f acc[4] = {};
    for (int kk = 0; kk < FDIM; kk += 32) {
        __builtin_prefetch(Ap + kk + 32, 0, 0);
        v16bf a;
#pragma unroll
        for (int t = 0; t < 16; ++t) a[t] = Ap[kk + kA(t, lh)];
#pragma unroll
        for (int nt = 0; nt < 4; ++nt) {
            const __bf16* Bp = kh + ((size_t)(h * N_ATOM + atoms[nt])) * FDIM;
            v16bf bfrag;
#pragma unroll
            for (int t = 0; t < 16; ++t) bfrag[t] = Bp[kk + kB(t, lh)];
            acc[nt] = __builtin_amdgcn_wmma_f32_16x16x32_bf16(
                false, a, false, bfrag, (short)0, acc[nt], false, false);
        }
    }
#pragma unroll
    for (int nt = 0; nt < 4; ++nt) {
        int   eg   = eb + nt * 16 + l16;
        float bias = attn_bias[h * E_EDGE + edge_map_tab[eg]];
#pragma unroll
        for (int r = 0; r < 8; ++r) {
            int ng = mt * 16 + r + lh * 8;
            scores[((size_t)(h * N_ATOM + ng)) * E_EDGE + eg] = acc[nt][r] + bias;
        }
    }
}

// ---------------- segment softmax over edges (batch_index sorted), attn out as bf16
__global__ void seg_softmax(const float* __restrict__ scores,
                            const int* __restrict__ batch_index,
                            const int* __restrict__ edge_map_tab,
                            const float* __restrict__ envelope,
                            __bf16* __restrict__ attn) {
    __shared__ unsigned smax[NSEG];
    __shared__ float    fmx[NSEG];
    __shared__ float    ssum[NSEG];
    int hn = blockIdx.x;                       // h*N + n  (4096 blocks)
    const float* row = scores + (size_t)hn * E_EDGE;
    int t = threadIdx.x;
    if (t < NSEG) { smax[t] = 0u; ssum[t] = 0.0f; }
    __syncthreads();
    for (int e = t; e < E_EDGE; e += 256) {
        atomicMax(&smax[batch_index[e]], fenc(row[e]));
    }
    __syncthreads();
    if (t < NSEG) fmx[t] = fdec(smax[t]);
    __syncthreads();
    for (int e = t; e < E_EDGE; e += 256) {
        int   sg  = batch_index[e];
        float env = envelope[edge_map_tab[e]];
        atomicAdd(&ssum[sg], expf(row[e] - fmx[sg]) * env);
    }
    __syncthreads();
    for (int e = t; e < E_EDGE; e += 256) {
        int   sg  = batch_index[e];
        float env = envelope[edge_map_tab[e]];
        float a   = expf(row[e] - fmx[sg]) * env * env / (ssum[sg] + 1e-16f);
        attn[(size_t)hn * E_EDGE + e] = (__bf16)a;
    }
}

// ---------------- zero-fill for out_gemm accumulator buffer
__global__ void zero_f32(float* __restrict__ p, int n) {
    int i = blockIdx.x * 256 + threadIdx.x;
    if (i < n) p[i] = 0.0f;
}

// ---------------- output GEMM: per-wave 16(n) x 48(f) tile, K split 8 ways
__global__ void out_gemm(const __bf16* __restrict__ attn, const __bf16* __restrict__ vh,
                         const int* __restrict__ atom_index, float* __restrict__ outh) {
    int w    = blockIdx.x * 8 + (threadIdx.x >> 5);   // 12288 wave jobs
    int h    = w / (32 * 6 * KSPLIT);
    int rem  = w % (32 * 6 * KSPLIT);
    int mt   = rem / (6 * KSPLIT);
    int rem2 = rem % (6 * KSPLIT);
    int fg   = rem2 / KSPLIT;
    int ks   = rem2 % KSPLIT;
    int lane = threadIdx.x & 31;
    int lh   = lane >> 4;
    int l16  = lane & 15;

    const int kbeg = ks * (E_EDGE / KSPLIT);
    const int kend = kbeg + (E_EDGE / KSPLIT);

    const __bf16* Ap = attn + ((size_t)(h * N_ATOM + mt * 16 + l16)) * E_EDGE;
    v8f acc[3] = {};
    for (int kk = kbeg; kk < kend; kk += 32) {
        __builtin_prefetch(Ap + kk + 32, 0, 0);
        v16bf a;
#pragma unroll
        for (int t = 0; t < 16; ++t) a[t] = Ap[kk + kA(t, lh)];
        int at[16];
#pragma unroll
        for (int t = 0; t < 16; ++t) at[t] = atom_index[kk + kB(t, lh)];
#pragma unroll
        for (int nt = 0; nt < 3; ++nt) {
            int f = fg * 48 + nt * 16 + l16;
            v16bf bfrag;
#pragma unroll
            for (int t = 0; t < 16; ++t)
                bfrag[t] = vh[((size_t)(h * N_ATOM + at[t])) * FDIM + f];
            acc[nt] = __builtin_amdgcn_wmma_f32_16x16x32_bf16(
                false, a, false, bfrag, (short)0, acc[nt], false, false);
        }
    }
#pragma unroll
    for (int nt = 0; nt < 3; ++nt)
#pragma unroll
        for (int r = 0; r < 8; ++r)
            atomicAdd(&outh[((size_t)(h * N_ATOM + mt * 16 + r + lh * 8)) * FDIM +
                            fg * 48 + nt * 16 + l16],
                      acc[nt][r]);
}

// ---------------- equivariant layernorm: block per atom, thread per channel
__device__ __forceinline__ float block_sum(float v, float* red) {
    int t = threadIdx.x;
    red[t] = v; __syncthreads();
    for (int s = 128; s > 0; s >>= 1) {
        if (t < s) red[t] += red[t + s];
        __syncthreads();
    }
    float r = red[0]; __syncthreads();
    return r;
}

__global__ void eln_kernel(const float* __restrict__ outh, const float* __restrict__ gamma,
                           const float* __restrict__ beta, float* __restrict__ normed) {
    __shared__ float red[256];
    int n = blockIdx.x, c = threadIdx.x;
    int h = c >> 5, dd = c & 31;
    const float* base = outh + ((size_t)(h * N_ATOM + n)) * FDIM + dd;
    float x[S_SPH];
#pragma unroll
    for (int s = 0; s < S_SPH; ++s) x[s] = base[s * DHEAD];
    float mu = block_sum(x[0], red) * (1.0f / 256.0f);
    float dv = x[0] - mu;
    float va = block_sum(dv * dv, red) * (1.0f / 256.0f);
    float y0 = dv / sqrtf(va + 1e-7f) * gamma[c] + beta[c];
    float n1 = (x[1]*x[1] + x[2]*x[2] + x[3]*x[3]) * (1.0f / 3.0f);
    float r1 = sqrtf(block_sum(n1, red) * (1.0f / 256.0f) + 1e-7f);
    float n2 = (x[4]*x[4] + x[5]*x[5] + x[6]*x[6] + x[7]*x[7] + x[8]*x[8]) * 0.2f;
    float r2 = sqrtf(block_sum(n2, red) * (1.0f / 256.0f) + 1e-7f);
    float* orow = normed + (size_t)n * S_SPH * CH;
    orow[c] = y0;
    float g1 = gamma[CH + c], g2 = gamma[2 * CH + c];
#pragma unroll
    for (int s = 1; s < 4; ++s) orow[s * CH + c] = x[s] / r1 * g1;
#pragma unroll
    for (int s = 4; s < 9; ++s) orow[s * CH + c] = x[s] / r2 * g2;
}

// ---------------- final so3_linear with Wo: (N,S,CH) -> (N,S,CIN)
__global__ void out_linear(const float* __restrict__ normed, const float* __restrict__ Wo,
                           const float* __restrict__ bo, float* __restrict__ out) {
    __shared__ float xr[CH];
    int ns = blockIdx.x;          // n*S + s
    int o  = threadIdx.x;         // 0..127
    int s  = ns % S_SPH;
    int l  = (s >= 4) ? 2 : ((s >= 1) ? 1 : 0);
    xr[o]       = normed[(size_t)ns * CH + o];
    xr[o + 128] = normed[(size_t)ns * CH + o + 128];
    __syncthreads();
    const float* Wl = Wo + (size_t)l * CH * CIN;
    float acc = bo[o];
    for (int c = 0; c < CH; ++c) acc += xr[c] * Wl[c * CIN + o];
    out[(size_t)ns * CIN + o] = acc;
}

extern "C" void kernel_launch(void* const* d_in, const int* in_sizes, int n_in,
                              void* d_out, int out_size, void* d_ws, size_t ws_size,
                              hipStream_t stream) {
    const float* q        = (const float*)d_in[0];
    const float* k        = (const float*)d_in[1];
    const float* v        = (const float*)d_in[2];
    const float* envelope = (const float*)d_in[3];
    const float* attn_bias= (const float*)d_in[4];
    const int*   atom_idx = (const int*)d_in[5];
    const int*   batch_idx= (const int*)d_in[6];
    const int*   edge_map = (const int*)d_in[7];
    const float* Wq = (const float*)d_in[8];
    const float* bq = (const float*)d_in[9];
    const float* Wk = (const float*)d_in[10];
    const float* bk = (const float*)d_in[11];
    const float* Wv = (const float*)d_in[12];
    const float* bv = (const float*)d_in[13];
    const float* gamma = (const float*)d_in[14];
    const float* beta  = (const float*)d_in[15];
    const float* Wo = (const float*)d_in[16];
    const float* bo = (const float*)d_in[17];

    char* ws = (char*)d_ws;
    size_t off = 0;
    auto take = [&](size_t bytes) { char* p = ws + off; off = (off + bytes + 255) & ~(size_t)255; return p; };
    __bf16* qh_b   = (__bf16*)take((size_t)NHEAD * N_ATOM * FDIM * 2);
    __bf16* kh_b   = (__bf16*)take((size_t)NHEAD * N_ATOM * FDIM * 2);
    __bf16* vh_b   = (__bf16*)take((size_t)NHEAD * N_ATOM * FDIM * 2);
    float*  scores = (float*) take((size_t)NHEAD * N_ATOM * E_EDGE * 4);
    __bf16* attn_b = (__bf16*)take((size_t)NHEAD * N_ATOM * E_EDGE * 2);
    float*  outh   = (float*) take((size_t)NHEAD * N_ATOM * FDIM * 4);
    float*  normed = (float*) take((size_t)N_ATOM * S_SPH * CH * 4);

    const float scale = sqrtf((float)DHEAD / 3.0f) / (float)DHEAD;

    so3_in_linear<<<N_ATOM * S_SPH, 256, 0, stream>>>(q, Wq, bq, qh_b, scale);
    so3_in_linear<<<N_ATOM * S_SPH, 256, 0, stream>>>(k, Wk, bk, kh_b, 1.0f);
    so3_in_linear<<<N_ATOM * S_SPH, 256, 0, stream>>>(v, Wv, bv, vh_b, 1.0f);

    // 8 heads * 32 n-tiles * 128 e-strips = 32768 wave jobs / 8 waves per block
    scores_gemm<<<4096, 256, 0, stream>>>(qh_b, kh_b, atom_idx, edge_map, attn_bias, scores);

    seg_softmax<<<NHEAD * N_ATOM, 256, 0, stream>>>(scores, batch_idx, edge_map, envelope, attn_b);

    const int outh_n = NHEAD * N_ATOM * FDIM;
    zero_f32<<<(outh_n + 255) / 256, 256, 0, stream>>>(outh, outh_n);

    // 8 heads * 32 n-tiles * 6 f-groups * 8 k-splits = 12288 wave jobs / 8 per block
    out_gemm<<<1536, 256, 0, stream>>>(attn_b, vh_b, atom_idx, outh);

    eln_kernel<<<N_ATOM, 256, 0, stream>>>(outh, gamma, beta, normed);

    out_linear<<<N_ATOM * S_SPH, 128, 0, stream>>>(normed, Wo, bo, (float*)d_out);
}
